// AdvancedTransformerModel_76081050681692
// MI455X (gfx1250) — matrix operand order repaired and checked by reading, not compiled, not run on hardware
//
#include <hip/hip_runtime.h>
#include <math.h>

// Problem constants (match reference)
#define V_  8000
#define D_  768
#define H_  12
#define HD_ 64
#define FF_ 3072
#define L_  12
#define P_  64
#define B_  4
#define S_  512
#define BS_ (B_*S_)
#define EPS_ 1e-5f

typedef __attribute__((ext_vector_type(16))) __bf16 v16bf;
typedef __attribute__((ext_vector_type(8)))  float  v8f;

// ---------------------------------------------------------------------------
// Block-wide all-reduce (wave32: shuffle within wave, LDS across waves)
// ---------------------------------------------------------------------------
__device__ __forceinline__ float block_allreduce(float v, float* red, bool ismax) {
  #pragma unroll
  for (int off = 16; off > 0; off >>= 1) {
    float o = __shfl_xor(v, off);
    v = ismax ? fmaxf(v, o) : (v + o);
  }
  __syncthreads();                       // guard LDS reuse across calls
  if ((threadIdx.x & 31) == 0) red[threadIdx.x >> 5] = v;
  __syncthreads();
  int nw = blockDim.x >> 5;
  float t = red[0];
  for (int i = 1; i < nw; ++i) t = ismax ? fmaxf(t, red[i]) : (t + red[i]);
  return t;
}

// ---------------------------------------------------------------------------
// Embedding gather: x[b,s,:] = tok_embed[ids[b,s]] + pos_embed[s]
// ---------------------------------------------------------------------------
__global__ void __launch_bounds__(256)
embed_kernel(const int* __restrict__ ids, const float* __restrict__ tok,
             const float* __restrict__ pos, float* __restrict__ x) {
  long idx = (long)blockIdx.x * blockDim.x + threadIdx.x;
  if (idx >= (long)BS_ * D_) return;
  int d  = (int)(idx % D_);
  int bs = (int)(idx / D_);
  int s  = bs % S_;
  int id = ids[bs];
  x[idx] = tok[(long)id * D_ + d] + pos[(long)s * D_ + d];
}

// ---------------------------------------------------------------------------
// LayerNorm: one 256-thread block per row of length D_
// ---------------------------------------------------------------------------
__global__ void __launch_bounds__(256)
ln_kernel(const float* __restrict__ x, const float* __restrict__ g,
          const float* __restrict__ bta, float* __restrict__ y) {
  __shared__ float red[8];
  long row = blockIdx.x;
  const float* xr = x + row * D_;
  float* yr = y + row * D_;
  float s = 0.f;
  for (int i = threadIdx.x; i < D_; i += 256) s += xr[i];
  float m = block_allreduce(s, red, false) * (1.f / D_);
  float v = 0.f;
  for (int i = threadIdx.x; i < D_; i += 256) { float d = xr[i] - m; v += d * d; }
  float var = block_allreduce(v, red, false) * (1.f / D_);
  float inv = rsqrtf(var + EPS_);
  for (int i = threadIdx.x; i < D_; i += 256)
    yr[i] = (xr[i] - m) * inv * g[i] + bta[i];
}

// ---------------------------------------------------------------------------
// Generic WMMA GEMM:
//   C[z][m][n] (acc?+=:=) act( scale * sum_k A[z][m][k]*B(k,n) + bias[n] )
//   B(k,n) = bT ? Bsrc[z][k][n] : Bsrc[z][n][k]
// 256 threads = 8 waves; block tile 64(M) x 32(N); waves 4x2 of 16x16 tiles.
// K-loop step 32, v_wmma_f32_16x16x32_bf16 with fp32 accumulation.
// Requires M%16==0, N%16==0, K%32==0 (true for all launches here).
// ---------------------------------------------------------------------------
__global__ void __launch_bounds__(256)
wmma_gemm(const float* __restrict__ A, long sAb, int lda,
          const float* __restrict__ Bsrc, long sBb, int ldb, int bT,
          const float* __restrict__ bias,
          float* __restrict__ C, long sCb, int ldc,
          int M, int N, int K, float scale, int act, int acc) {
  int batch = blockIdx.z;
  A    += (long)batch * sAb;
  Bsrc += (long)batch * sBb;
  C    += (long)batch * sCb;

  int wave = threadIdx.x >> 5;
  int lane = threadIdx.x & 31;
  int wm = wave & 3, wn = wave >> 2;
  int m0 = blockIdx.y * 64 + wm * 16;
  int n0 = blockIdx.x * 32 + wn * 16;
  if (m0 >= M || n0 >= N) return;     // wave-uniform

  int lo = lane & 15;                  // 0..15
  int hi = lane >> 4;                  // 0 or 1

  v8f c = {};
  int arow    = m0 + lo;               // A-frag: this lane's M row
  int bk_lane = lo + 16 * hi;

  for (int k0 = 0; k0 < K; k0 += 32) {
    v16bf a, b;
    // A 16x32 bf16 layout: lanes 0-15 K={0-7,16-23}, lanes 16-31 K={8-15,24-31}
    const float* Ar = A + (long)arow * lda + k0 + 8 * hi;
    #pragma unroll
    for (int e = 0; e < 8; ++e) {
      a[e]     = (__bf16)Ar[e];
      a[e + 8] = (__bf16)Ar[e + 16];
    }
    // B 32x16 bf16 layout: lane holds N=n0+lo, K same as A-frag
    int kb = k0 + 8 * hi;
    if (bT) {
      const float* Br = Bsrc + (long)kb * ldb + n0 + lo;
      #pragma unroll
      for (int e = 0; e < 8; ++e) {
        b[e]     = (__bf16)Br[(long)e * ldb];
        b[e + 8] = (__bf16)Br[(long)(e + 16) * ldb];
      }
    } else {
      const float* Br = Bsrc + (long)(n0 + lo) * ldb + kb;
      #pragma unroll
      for (int e = 0; e < 8; ++e) {
        b[e]     = (__bf16)Br[e];
        b[e + 8] = (__bf16)Br[e + 16];
      }
    }
    c = __builtin_amdgcn_wmma_f32_16x16x32_bf16(
        /*neg_a=*/false, a, /*neg_b=*/false, b,
        /*c_mod=*/(short)0, c, /*reuse_a=*/false, /*reuse_b=*/false);
  }

  // Epilogue. C/D layout: element e -> M = m0 + e + 8*hi, N = n0 + (lane&15)
  int cn = n0 + lo;
  float bv = bias ? bias[cn] : 0.f;
  #pragma unroll
  for (int e = 0; e < 8; ++e) {
    int cm = m0 + e + 8 * hi;
    float r = scale * c[e] + bv;
    if (act == 1)      r = tanhf(r);
    else if (act == 2) r = 0.5f * r * (1.f + erff(r * 0.70710678f));  // exact GELU
    float* p = C + (long)cm * ldc + cn;
    if (acc) *p += r; else *p = r;
  }
}

// ---------------------------------------------------------------------------
// Head-averaged softmax: aw[b,q,:] = mean_h softmax_k(scores[b,h,q,:])
// One 256-thread block per (b,q); each thread owns columns t and t+256.
// ---------------------------------------------------------------------------
__global__ void __launch_bounds__(256)
attn_softmax_mean(const float* __restrict__ scores, float* __restrict__ aw) {
  __shared__ float red[8];
  long bq = blockIdx.x;
  long b = bq / S_, q = bq % S_;
  float a0 = 0.f, a1 = 0.f;
  for (int h = 0; h < H_; ++h) {
    const float* row = scores + ((b * H_ + h) * S_ + q) * S_;
    float v0 = row[threadIdx.x];
    float v1 = row[threadIdx.x + 256];
    float mx = block_allreduce(fmaxf(v0, v1), red, true);
    float e0 = expf(v0 - mx), e1 = expf(v1 - mx);
    float sm = block_allreduce(e0 + e1, red, false);
    float inv = 1.f / sm;
    a0 += e0 * inv; a1 += e1 * inv;
  }
  float* awr = aw + bq * S_;
  awr[threadIdx.x]       = a0 * (1.f / H_);
  awr[threadIdx.x + 256] = a1 * (1.f / H_);
}

// ---------------------------------------------------------------------------
// Blended softmax: bw = softmax_k( log(aw+1e-6) + alpha * (pc_raw+1)*0.5 )
// ---------------------------------------------------------------------------
__global__ void __launch_bounds__(256)
blend_softmax(const float* __restrict__ aw, const float* __restrict__ pc,
              const float* __restrict__ alpha_p, float* __restrict__ bw) {
  __shared__ float red[8];
  long bq = blockIdx.x;
  float alpha = alpha_p[0];
  const float* awr = aw + bq * S_;
  const float* pcr = pc + bq * S_;
  float l0 = logf(awr[threadIdx.x]       + 1e-6f) + alpha * ((pcr[threadIdx.x]       + 1.f) * 0.5f);
  float l1 = logf(awr[threadIdx.x + 256] + 1e-6f) + alpha * ((pcr[threadIdx.x + 256] + 1.f) * 0.5f);
  float mx = block_allreduce(fmaxf(l0, l1), red, true);
  float e0 = expf(l0 - mx), e1 = expf(l1 - mx);
  float sm = block_allreduce(e0 + e1, red, false);
  float inv = 1.f / sm;
  float* bwr = bw + bq * S_;
  bwr[threadIdx.x]       = e0 * inv;
  bwr[threadIdx.x + 256] = e1 * inv;
}

// ---------------------------------------------------------------------------
// Host orchestration
// ---------------------------------------------------------------------------
extern "C" void kernel_launch(void* const* d_in, const int* in_sizes, int n_in,
                              void* d_out, int out_size, void* d_ws, size_t ws_size,
                              hipStream_t stream) {
  (void)in_sizes; (void)n_in; (void)out_size; (void)ws_size;

  const int*   ids   = (const int*)  d_in[0];
  const float* tok   = (const float*)d_in[1];
  const float* pos   = (const float*)d_in[2];
  const float* ln1g  = (const float*)d_in[3];
  const float* ln1b  = (const float*)d_in[4];
  const float* w_in  = (const float*)d_in[5];
  const float* b_in  = (const float*)d_in[6];
  const float* w_ph  = (const float*)d_in[7];
  const float* alpha = (const float*)d_in[8];
  const float* ln2g  = (const float*)d_in[9];
  const float* ln2b  = (const float*)d_in[10];
  const float* w_f1  = (const float*)d_in[11];
  const float* b_f1  = (const float*)d_in[12];
  const float* w_f2  = (const float*)d_in[13];
  const float* b_f2  = (const float*)d_in[14];
  const float* lnfg  = (const float*)d_in[15];
  const float* lnfb  = (const float*)d_in[16];
  const float* w_out = (const float*)d_in[17];
  const float* b_out = (const float*)d_in[18];
  float* out = (float*)d_out;

  // Workspace layout (fp32), every buffer fully written before read each call
  float* ws  = (float*)d_ws;
  float* x   = ws;                        // [BS, D]
  float* xn  = x   + (long)BS_ * D_;      // [BS, D]
  float* qkv = xn  + (long)BS_ * D_;      // [BS, 3D]
  float* pv  = qkv + (long)BS_ * 3 * D_;  // [BS, P]
  float* aw  = pv  + (long)BS_ * P_;      // [B, S, S]
  float* pc  = aw  + (long)B_ * S_ * S_;  // [B, S, S]
  float* bw  = pc  + (long)B_ * S_ * S_;  // [B, S, S]
  float* sc  = bw  + (long)B_ * S_ * S_;  // [B, H, S, S] (largest)
  float* h1  = sc;                        // [BS, FF] reuses score region

  dim3 blk(256);

  embed_kernel<<<(BS_ * D_ + 255) / 256, blk, 0, stream>>>(ids, tok, pos, x);

  for (int l = 0; l < L_; ++l) {
    // xn = LN1(x)
    ln_kernel<<<BS_, blk, 0, stream>>>(x, ln1g + (long)l * D_, ln1b + (long)l * D_, xn);

    // qkv = xn @ w_in^T + b_in            [2048 x 2304 x 768]
    wmma_gemm<<<dim3(3 * D_ / 32, BS_ / 64, 1), blk, 0, stream>>>(
        xn, 0, D_, w_in + (long)l * 3 * D_ * D_, 0, D_, 0,
        b_in + (long)l * 3 * D_, qkv, 0, 3 * D_,
        BS_, 3 * D_, D_, 1.f, 0, 0);

    // scores[b,h] = (Q_h @ K_h^T) / 8     [512 x 512 x 64] per (b,h)
    for (int h = 0; h < H_; ++h) {
      wmma_gemm<<<dim3(S_ / 32, S_ / 64, B_), blk, 0, stream>>>(
          qkv + h * HD_,       (long)S_ * 3 * D_, 3 * D_,
          qkv + D_ + h * HD_,  (long)S_ * 3 * D_, 3 * D_, 0,
          nullptr,
          sc + (long)h * S_ * S_, (long)H_ * S_ * S_, S_,
          S_, S_, HD_, 0.125f, 0, 0);
    }

    // aw = mean_h softmax(scores)
    attn_softmax_mean<<<B_ * S_, blk, 0, stream>>>(sc, aw);

    // pv = tanh(xn @ w_phase^T)           [2048 x 64 x 768]
    wmma_gemm<<<dim3(P_ / 32, BS_ / 64, 1), blk, 0, stream>>>(
        xn, 0, D_, w_ph + (long)l * P_ * D_, 0, D_, 0,
        nullptr, pv, 0, P_, BS_, P_, D_, 1.f, 1, 0);

    // pc_raw[b] = pv[b] @ pv[b]^T         [512 x 512 x 64] per b
    wmma_gemm<<<dim3(S_ / 32, S_ / 64, B_), blk, 0, stream>>>(
        pv, (long)S_ * P_, P_, pv, (long)S_ * P_, P_, 0,
        nullptr, pc, (long)S_ * S_, S_, S_, S_, P_, 1.f, 0, 0);

    // bw = softmax(log(aw+1e-6) + alpha*(pc+1)/2)
    blend_softmax<<<B_ * S_, blk, 0, stream>>>(aw, pc, alpha + l, bw);

    // x += bw[b] @ xn[b]                  [512 x 768 x 512] per b (bT=1)
    wmma_gemm<<<dim3(D_ / 32, S_ / 64, B_), blk, 0, stream>>>(
        bw, (long)S_ * S_, S_, xn, (long)S_ * D_, D_, 1,
        nullptr, x, (long)S_ * D_, D_, S_, D_, S_, 1.f, 0, 1);

    // xn = LN2(x)
    ln_kernel<<<BS_, blk, 0, stream>>>(x, ln2g + (long)l * D_, ln2b + (long)l * D_, xn);

    // h1 = gelu(xn @ w_ff1^T + b_ff1)     [2048 x 3072 x 768]
    wmma_gemm<<<dim3(FF_ / 32, BS_ / 64, 1), blk, 0, stream>>>(
        xn, 0, D_, w_f1 + (long)l * FF_ * D_, 0, D_, 0,
        b_f1 + (long)l * FF_, h1, 0, FF_, BS_, FF_, D_, 1.f, 2, 0);

    // x += h1 @ w_ff2^T + b_ff2           [2048 x 768 x 3072]
    wmma_gemm<<<dim3(D_ / 32, BS_ / 64, 1), blk, 0, stream>>>(
        h1, 0, FF_, w_f2 + (long)l * D_ * FF_, 0, FF_, 0,
        b_f2 + (long)l * D_, x, 0, D_, BS_, D_, FF_, 1.f, 0, 1);
  }

  // Final LN + vocab projection          [2048 x 8000 x 768]
  ln_kernel<<<BS_, blk, 0, stream>>>(x, lnfg, lnfb, xn);
  wmma_gemm<<<dim3(V_ / 32, BS_ / 64, 1), blk, 0, stream>>>(
      xn, 0, D_, w_out, 0, D_, 0, b_out, out, 0, V_,
      BS_, V_, D_, 1.f, 0, 0);
}
